// RadixMLPQwen3MLP_4105988735673
// MI455X (gfx1250) — compile-verified, matrix operand order checked by reading, not run on hardware
//
#include <hip/hip_runtime.h>
#include <hip/hip_bf16.h>

// ---------------------------------------------------------------------------
// RadixMLP (gather -> SwiGLU MLP -> scatter) for gfx1250 / MI455X.
// bf16 WMMA (v_wmma_f32_16x16x32_bf16), fp32 accumulate,
// double-buffered LDS with async global->LDS staging (ASYNCcnt).
// ---------------------------------------------------------------------------

typedef __attribute__((ext_vector_type(16))) __bf16 v16bf;
typedef __attribute__((ext_vector_type(8)))  __bf16 bf16x8;
typedef __attribute__((ext_vector_type(4)))  __bf16 bf16x4;
typedef __attribute__((ext_vector_type(8)))  float  v8f;
typedef __attribute__((ext_vector_type(4)))  float  f32x4;
typedef __attribute__((ext_vector_type(4)))  int    i32x4;

#define NUM_TOKENS 16384
#define COMPACT    8192
#define HIDDEN     2048
#define INTER      8192

#if defined(__HIP_DEVICE_COMPILE__) && \
    __has_builtin(__builtin_amdgcn_global_load_async_to_lds_b128)
#define HAVE_ASYNC_LDS 1
#else
#define HAVE_ASYNC_LDS 0
#endif

// round-to-nearest-even fp32 -> bf16
__device__ __forceinline__ __bf16 f2bf(float f) {
  unsigned u = __builtin_bit_cast(unsigned, f);
  u += 0x7FFFu + ((u >> 16) & 1u);
  unsigned short s = (unsigned short)(u >> 16);
  return __builtin_bit_cast(__bf16, s);
}

// 16-byte global -> LDS copy: async on gfx1250 (GLOBAL_LOAD_ASYNC_TO_LDS_B128,
// tracked by ASYNCcnt, no VGPR round trip), else sync fallback.
__device__ __forceinline__ void cp_lds16(__bf16* dst, const __bf16* src) {
#if HAVE_ASYNC_LDS
  typedef __attribute__((address_space(1))) i32x4 gvec_t;
  typedef __attribute__((address_space(3))) i32x4 lvec_t;
  __builtin_amdgcn_global_load_async_to_lds_b128(
      (gvec_t*)src, (lvec_t*)dst, 0, 0);
#else
  *(bf16x8*)dst = *(const bf16x8*)src;
#endif
}

__device__ __forceinline__ void wait_async0() {
#if HAVE_ASYNC_LDS
#if __has_builtin(__builtin_amdgcn_s_wait_asynccnt)
  __builtin_amdgcn_s_wait_asynccnt(0);
#else
  asm volatile("s_wait_asynccnt 0x0" ::: "memory");
#endif
#endif
}

// Load one 16x32 bf16 WMMA operand fragment from an LDS tile.
// Per ISA 16-bit A layout: lanes 0-15 hold K {0..7,16..23}, lanes 16-31 hold
// K {8..15,24..31}; each lane = row (lane&15). Two 16B LDS loads per lane.
__device__ __forceinline__ v16bf ld_frag(const __bf16* p, int lane, int ldm) {
  const int r  = lane & 15;
  const int ko = (lane >> 4) * 8;
  bf16x8 lo = *(const bf16x8*)(p + r * ldm + ko);
  bf16x8 hi = *(const bf16x8*)(p + r * ldm + ko + 16);
  return __builtin_shufflevector(lo, hi,
      0,1,2,3,4,5,6,7,8,9,10,11,12,13,14,15);
}

// ---------------------------------------------------------------------------
// Elementwise prologue / epilogue kernels
// ---------------------------------------------------------------------------

__global__ __launch_bounds__(256) void gather_cvt_kernel(
    const float* __restrict__ x, const int* __restrict__ fg,
    __bf16* __restrict__ xc) {
  const int H4 = HIDDEN / 4;
  long i = (long)blockIdx.x * blockDim.x + threadIdx.x;  // over COMPACT*H4
  long c = i / H4;
  int  h4 = (int)(i - c * H4);
  if (c >= COMPACT) return;
  long src = (long)fg[c] * H4 + h4;
  f32x4 v = ((const f32x4*)x)[src];
  bf16x4 o;
  o[0] = f2bf(v[0]); o[1] = f2bf(v[1]); o[2] = f2bf(v[2]); o[3] = f2bf(v[3]);
  ((bf16x4*)xc)[i] = o;
}

__global__ __launch_bounds__(256) void cvt_kernel(
    const float* __restrict__ in, __bf16* __restrict__ out, long n4) {
  long i = (long)blockIdx.x * blockDim.x + threadIdx.x;
  if (i >= n4) return;
  f32x4 v = ((const f32x4*)in)[i];
  bf16x4 o;
  o[0] = f2bf(v[0]); o[1] = f2bf(v[1]); o[2] = f2bf(v[2]); o[3] = f2bf(v[3]);
  ((bf16x4*)out)[i] = o;
}

__global__ __launch_bounds__(256) void scatter_kernel(
    const float* __restrict__ down, const int* __restrict__ sc,
    float* __restrict__ out) {
  const int H4 = HIDDEN / 4;
  long i = (long)blockIdx.x * blockDim.x + threadIdx.x;  // over NUM_TOKENS*H4
  long n = i / H4;
  int  h4 = (int)(i - n * H4);
  if (n >= NUM_TOKENS) return;
  ((f32x4*)out)[i] = ((const f32x4*)down)[(long)sc[n] * H4 + h4];
}

// ---------------------------------------------------------------------------
// GEMM1: act[C,I] = silu(Xc @ Wg^T) * (Xc @ Wu^T)   (bf16 in, bf16 out)
// Block 128x128, K-tile 64, 8 waves of 64x32, dual accumulators (gate & up),
// double-buffered LDS + async staging.
// ---------------------------------------------------------------------------
__global__ __launch_bounds__(256, 1) void gemm_gate_up_kernel(
    const __bf16* __restrict__ Xc, const __bf16* __restrict__ Wg,
    const __bf16* __restrict__ Wu, __bf16* __restrict__ Act) {
  constexpr int K   = HIDDEN;  // 2048
  constexpr int N   = INTER;   // 8192
  constexpr int LDT = 72;      // padded LDS stride (bf16)

  __shared__ __align__(16) __bf16 As [2][128 * LDT];
  __shared__ __align__(16) __bf16 Bgs[2][128 * LDT];
  __shared__ __align__(16) __bf16 Bus[2][128 * LDT];

  const int t    = threadIdx.x;
  const int lane = t & 31;
  const int wid  = t >> 5;
  const int wm   = wid & 1;   // wave row: 2 x 64 rows
  const int wn   = wid >> 1;  // wave col: 4 x 32 cols
  const long m0 = (long)blockIdx.y * 128;
  const long n0 = (long)blockIdx.x * 128;

  v8f accg[4][2], accu[4][2];
#pragma unroll
  for (int mt = 0; mt < 4; ++mt)
#pragma unroll
    for (int nt = 0; nt < 2; ++nt) { accg[mt][nt] = (v8f){}; accu[mt][nt] = (v8f){}; }

  auto stage = [&](int buf, int kt) {
#pragma unroll
    for (int i = 0; i < 4; ++i) {
      int s   = t + 256 * i;       // 0..1023 slots of 8 bf16
      int row = s >> 3;
      int col = (s & 7) * 8;
      cp_lds16(&As [buf][row * LDT + col], &Xc[(m0 + row) * K + kt + col]);
      cp_lds16(&Bgs[buf][row * LDT + col], &Wg[(n0 + row) * K + kt + col]);
      cp_lds16(&Bus[buf][row * LDT + col], &Wu[(n0 + row) * K + kt + col]);
    }
  };

  stage(0, 0);
  for (int kt = 0; kt < K; kt += 64) {
    const int cur = (kt >> 6) & 1;
    wait_async0();
    __syncthreads();                       // buffer `cur` ready for all waves
    if (kt + 64 < K) stage(cur ^ 1, kt + 64);  // overlap next tile with compute

#pragma unroll
    for (int k0 = 0; k0 < 64; k0 += 32) {
      v16bf a[4], bg[2], bu[2];
#pragma unroll
      for (int mt = 0; mt < 4; ++mt)
        a[mt] = ld_frag(&As[cur][(wm * 64 + mt * 16) * LDT + k0], lane, LDT);
#pragma unroll
      for (int nt = 0; nt < 2; ++nt) {
        bg[nt] = ld_frag(&Bgs[cur][(wn * 32 + nt * 16) * LDT + k0], lane, LDT);
        bu[nt] = ld_frag(&Bus[cur][(wn * 32 + nt * 16) * LDT + k0], lane, LDT);
      }
#pragma unroll
      for (int mt = 0; mt < 4; ++mt)
#pragma unroll
        for (int nt = 0; nt < 2; ++nt) {
          accg[mt][nt] = __builtin_amdgcn_wmma_f32_16x16x32_bf16(
              false, a[mt], false, bg[nt], (short)0, accg[mt][nt], false, false);
          accu[mt][nt] = __builtin_amdgcn_wmma_f32_16x16x32_bf16(
              false, a[mt], false, bu[nt], (short)0, accu[mt][nt], false, false);
        }
    }
  }

  // epilogue: act = silu(gate) * up -> bf16
#pragma unroll
  for (int mt = 0; mt < 4; ++mt)
#pragma unroll
    for (int nt = 0; nt < 2; ++nt) {
      long mb = m0 + wm * 64 + mt * 16 + ((lane >> 4) * 8);
      long nb = n0 + wn * 32 + nt * 16 + (lane & 15);
      v8f g = accg[mt][nt], u = accu[mt][nt];
#pragma unroll
      for (int r = 0; r < 8; ++r) {
        float gv = g[r];
        float av = (gv / (1.0f + __expf(-gv))) * u[r];
        Act[(mb + r) * (long)N + nb] = f2bf(av);
      }
    }
}

// ---------------------------------------------------------------------------
// GEMM2: down[C,H] = Act @ Wd^T   (bf16 in, fp32 out)
// ---------------------------------------------------------------------------
__global__ __launch_bounds__(256, 1) void gemm_down_kernel(
    const __bf16* __restrict__ Act, const __bf16* __restrict__ Wd,
    float* __restrict__ Down) {
  constexpr int K   = INTER;   // 8192
  constexpr int N   = HIDDEN;  // 2048
  constexpr int LDT = 72;

  __shared__ __align__(16) __bf16 As[2][128 * LDT];
  __shared__ __align__(16) __bf16 Bs[2][128 * LDT];

  const int t    = threadIdx.x;
  const int lane = t & 31;
  const int wid  = t >> 5;
  const int wm   = wid & 1;
  const int wn   = wid >> 1;
  const long m0 = (long)blockIdx.y * 128;
  const long n0 = (long)blockIdx.x * 128;

  v8f acc[4][2];
#pragma unroll
  for (int mt = 0; mt < 4; ++mt)
#pragma unroll
    for (int nt = 0; nt < 2; ++nt) acc[mt][nt] = (v8f){};

  auto stage = [&](int buf, int kt) {
#pragma unroll
    for (int i = 0; i < 4; ++i) {
      int s   = t + 256 * i;
      int row = s >> 3;
      int col = (s & 7) * 8;
      cp_lds16(&As[buf][row * LDT + col], &Act[(m0 + row) * (long)K + kt + col]);
      cp_lds16(&Bs[buf][row * LDT + col], &Wd [(n0 + row) * (long)K + kt + col]);
    }
  };

  stage(0, 0);
  for (int kt = 0; kt < K; kt += 64) {
    const int cur = (kt >> 6) & 1;
    wait_async0();
    __syncthreads();
    if (kt + 64 < K) stage(cur ^ 1, kt + 64);

#pragma unroll
    for (int k0 = 0; k0 < 64; k0 += 32) {
      v16bf a[4], b[2];
#pragma unroll
      for (int mt = 0; mt < 4; ++mt)
        a[mt] = ld_frag(&As[cur][(wm * 64 + mt * 16) * LDT + k0], lane, LDT);
#pragma unroll
      for (int nt = 0; nt < 2; ++nt)
        b[nt] = ld_frag(&Bs[cur][(wn * 32 + nt * 16) * LDT + k0], lane, LDT);
#pragma unroll
      for (int mt = 0; mt < 4; ++mt)
#pragma unroll
        for (int nt = 0; nt < 2; ++nt)
          acc[mt][nt] = __builtin_amdgcn_wmma_f32_16x16x32_bf16(
              false, a[mt], false, b[nt], (short)0, acc[mt][nt], false, false);
    }
  }

#pragma unroll
  for (int mt = 0; mt < 4; ++mt)
#pragma unroll
    for (int nt = 0; nt < 2; ++nt) {
      long mb = m0 + wm * 64 + mt * 16 + ((lane >> 4) * 8);
      long nb = n0 + wn * 32 + nt * 16 + (lane & 15);
      v8f c = acc[mt][nt];
#pragma unroll
      for (int r = 0; r < 8; ++r)
        Down[(mb + r) * (long)N + nb] = c[r];
    }
}

// ---------------------------------------------------------------------------
// Launch
// ---------------------------------------------------------------------------
extern "C" void kernel_launch(void* const* d_in, const int* in_sizes, int n_in,
                              void* d_out, int out_size, void* d_ws, size_t ws_size,
                              hipStream_t stream) {
  (void)in_sizes; (void)n_in; (void)out_size; (void)ws_size;

  const float* x      = (const float*)d_in[0];  // [N, H]
  const float* W_gate = (const float*)d_in[1];  // [I, H]
  const float* W_up   = (const float*)d_in[2];  // [I, H]
  const float* W_down = (const float*)d_in[3];  // [H, I]
  const int*   fg     = (const int*)d_in[4];    // [C]
  const int*   sc     = (const int*)d_in[5];    // [N]
  float*       out    = (float*)d_out;          // [N, H]

  // workspace layout (bytes)
  char* ws = (char*)d_ws;
  const size_t SZ_WH = (size_t)INTER * HIDDEN * 2;      // 32MB (bf16 weight)
  __bf16* xc_bf  = (__bf16*)(ws);                               // 32MB
  __bf16* wg_bf  = (__bf16*)(ws + 1 * SZ_WH);                   // 32MB
  __bf16* wu_bf  = (__bf16*)(ws + 2 * SZ_WH);                   // 32MB
  __bf16* wd_bf  = (__bf16*)(ws + 3 * SZ_WH);                   // 32MB
  __bf16* act_bf = (__bf16*)(ws + 4 * SZ_WH);                   // 128MB
  float*  down   = (float*)(ws + 4 * SZ_WH + (size_t)COMPACT * INTER * 2);  // 64MB

  // 1) gather + convert x, convert weights
  {
    long n4 = (long)COMPACT * (HIDDEN / 4);
    gather_cvt_kernel<<<(n4 + 255) / 256, 256, 0, stream>>>(x, fg, xc_bf);
    long w4 = (long)INTER * HIDDEN / 4;
    cvt_kernel<<<(w4 + 255) / 256, 256, 0, stream>>>(W_gate, wg_bf, w4);
    cvt_kernel<<<(w4 + 255) / 256, 256, 0, stream>>>(W_up,   wu_bf, w4);
    cvt_kernel<<<(w4 + 255) / 256, 256, 0, stream>>>(W_down, wd_bf, w4);
  }

  // 2) GEMM1 fused gate/up + SiLU*up
  {
    dim3 grid(INTER / 128, COMPACT / 128);  // (64, 64)
    gemm_gate_up_kernel<<<grid, 256, 0, stream>>>(xc_bf, wg_bf, wu_bf, act_bf);
  }

  // 3) GEMM2 down-projection
  {
    dim3 grid(HIDDEN / 128, COMPACT / 128);  // (16, 64)
    gemm_down_kernel<<<grid, 256, 0, stream>>>(act_bf, wd_bf, down);
  }

  // 4) scatter back to token slots
  {
    long n4 = (long)NUM_TOKENS * (HIDDEN / 4);
    scatter_kernel<<<(n4 + 255) / 256, 256, 0, stream>>>(down, sc, out);
  }
}
